// GCN_34660386078859
// MI455X (gfx1250) — compile-verified, matrix operand order checked by reading, not run on hardware
//
#include <hip/hip_runtime.h>
#include <hip/hip_bf16.h>

typedef __attribute__((ext_vector_type(16))) _Float16 v16h;
typedef __attribute__((ext_vector_type(8)))  float    v8f;

#define IN_DIM 128   // K of both GEMMs (hidden dim == in dim == 128)

// ---------------------------------------------------------------------------
// Pack a KxN f32 weight matrix (K=128) into the WMMA 16-bit B-fragment layout,
// converted to f16. Layout per (n-tile t, k-step kk, lane l):
//   half = l>>4, r = l&15, n = t*16 + r, k0 = kk*32
//   dst[j]   = W[k0 + half*8      + j][n]   j=0..7
//   dst[8+j] = W[k0 + 16 + half*8 + j][n]   j=0..7
// so each wave's B operand is 16 contiguous halves per lane (2 x b128 loads).
// ---------------------------------------------------------------------------
__global__ void gcn_pack_w(const float* __restrict__ W, _Float16* __restrict__ Wp,
                           int Nt /* N/16 */) {
    int id = blockIdx.x * blockDim.x + threadIdx.x;
    int total = Nt * 4 * 32;
    if (id >= total) return;
    int lane = id & 31;
    int kk   = (id >> 5) & 3;
    int t    = id >> 7;
    int half = lane >> 4;
    int r    = lane & 15;
    int N    = Nt * 16;
    int n    = t * 16 + r;
    int k0   = kk * 32;
    _Float16* dst = Wp + (size_t)id * 16;
#pragma unroll
    for (int j = 0; j < 8; ++j)
        dst[j] = (_Float16)W[(size_t)(k0 + half * 8 + j) * N + n];
#pragma unroll
    for (int j = 0; j < 8; ++j)
        dst[8 + j] = (_Float16)W[(size_t)(k0 + 16 + half * 8 + j) * N + n];
}

// ---------------------------------------------------------------------------
// SpMM scatter: one wave per edge. Each lane loads float4 of the gathered row
// (128-bit global load) and does 4 global f32 atomic adds into the accumulator.
// x and agg both fit in the 192MB L2, so this runs at L2 bandwidth.
// ---------------------------------------------------------------------------
__global__ void gcn_spmm(const int* __restrict__ row, const int* __restrict__ col,
                         const float* __restrict__ vals, const float* __restrict__ x,
                         float* __restrict__ agg, int nEdges) {
    int gid  = blockIdx.x * blockDim.x + threadIdx.x;
    int e    = gid >> 5;
    int lane = gid & 31;
    if (e >= nEdges) return;
    int   r = row[e];
    int   c = col[e];
    float v = vals[e];
    const float4* xp = (const float4*)(x + (size_t)c * IN_DIM);
    float4 m = xp[lane];                       // 32 lanes x 4 = 128 dims
    float* op = agg + (size_t)r * IN_DIM + lane * 4;
    atomicAdd(op + 0, v * m.x);
    atomicAdd(op + 1, v * m.y);
    atomicAdd(op + 2, v * m.z);
    atomicAdd(op + 3, v * m.w);
}

// ---------------------------------------------------------------------------
// WMMA GEMM: C[M,N] = A[M,128] @ B[128,N], N = NTILES*16, optional relu.
// One wave computes a full 16 x N strip: the A fragment for each k-step is
// built once (two float4 loads + cvt per half) and reused against NTILES
// pre-packed B fragments; NTILES v8f accumulators live in VGPRs
// (NTILES=8 -> 64 acc VGPRs, fine under wave32 budget).
// K-loop = 4 k-steps of v_wmma_f32_16x16x32_f16 per n-tile.
// ---------------------------------------------------------------------------
template <int NTILES, bool RELU>
__global__ void gcn_gemm_wmma(const float* __restrict__ A,
                              const _Float16* __restrict__ Bp,
                              float* __restrict__ C, int M) {
    const int N = NTILES * 16;
    int lane = threadIdx.x & 31;
    int warp = threadIdx.x >> 5;
    int wavesPerBlock = blockDim.x >> 5;
    int mt = blockIdx.x * wavesPerBlock + warp;    // m-tile index, one per wave
    int mtiles = M >> 4;
    if (mt >= mtiles) return;          // uniform per-wave branch; EXEC stays all-1s

    int half = lane >> 4;
    int r    = lane & 15;

    int mrow = mt * 16 + r;            // A-fragment row for this lane (M%16==0)

    v8f acc[NTILES];
#pragma unroll
    for (int t = 0; t < NTILES; ++t) acc[t] = (v8f){};

#pragma unroll
    for (int kk = 0; kk < 4; ++kk) {
        int k0 = kk * 32;
        const float* ap = A + (size_t)mrow * IN_DIM + k0 + half * 8;
        float4 f0 = *(const float4*)(ap + 0);
        float4 f1 = *(const float4*)(ap + 4);
        float4 f2 = *(const float4*)(ap + 16);
        float4 f3 = *(const float4*)(ap + 20);
        v16h a;
        a[0]  = (_Float16)f0.x; a[1]  = (_Float16)f0.y; a[2]  = (_Float16)f0.z; a[3]  = (_Float16)f0.w;
        a[4]  = (_Float16)f1.x; a[5]  = (_Float16)f1.y; a[6]  = (_Float16)f1.z; a[7]  = (_Float16)f1.w;
        a[8]  = (_Float16)f2.x; a[9]  = (_Float16)f2.y; a[10] = (_Float16)f2.z; a[11] = (_Float16)f2.w;
        a[12] = (_Float16)f3.x; a[13] = (_Float16)f3.y; a[14] = (_Float16)f3.z; a[15] = (_Float16)f3.w;

#pragma unroll
        for (int t = 0; t < NTILES; ++t) {
            v16h b = *(const v16h*)(Bp + (((size_t)t * 4 + kk) * 32 + lane) * 16);
            acc[t] = __builtin_amdgcn_wmma_f32_16x16x32_f16(
                        /*neg_a=*/false, a, /*neg_b=*/false, b,
                        /*c_mod=*/(short)0, acc[t],
                        /*reuse_a=*/false, /*reuse_b=*/false);
        }
    }

    // D layout: VGPR v -> row = mt*16 + half*8 + v, col = t*16 + r
    int mbase = mt * 16 + half * 8;
#pragma unroll
    for (int t = 0; t < NTILES; ++t) {
        int ncol = t * 16 + r;
#pragma unroll
        for (int v = 0; v < 8; ++v) {
            float val = acc[t][v];
            if (RELU) val = fmaxf(val, 0.0f);
            C[(size_t)(mbase + v) * N + ncol] = val;
        }
    }
}

extern "C" void kernel_launch(void* const* d_in, const int* in_sizes, int n_in,
                              void* d_out, int out_size, void* d_ws, size_t ws_size,
                              hipStream_t stream) {
    const float* x    = (const float*)d_in[0];
    const int*   row  = (const int*)d_in[1];
    const int*   col  = (const int*)d_in[2];
    const float* vals = (const float*)d_in[3];
    const float* W1   = (const float*)d_in[4];
    const float* W2   = (const float*)d_in[5];
    float* out = (float*)d_out;

    const int NN = in_sizes[0] / IN_DIM;   // 50000 nodes
    const int NE = in_sizes[1];            // 800000 edges

    // Workspace layout
    float*     aggF = (float*)d_ws;                         // [NN,128] f32 (reused)
    float*     hF   = aggF + (size_t)NN * IN_DIM;           // [NN,128] f32
    _Float16*  W1p  = (_Float16*)(hF + (size_t)NN * IN_DIM);// 128*128 f16 packed
    _Float16*  W2p  = W1p + 128 * 128;                      // 128*64  f16 packed

    // 1) pack weights into WMMA fragment layout
    gcn_pack_w<<<4, 256, 0, stream>>>(W1, W1p, 8);  // N=128 -> 8 n-tiles
    gcn_pack_w<<<2, 256, 0, stream>>>(W2, W2p, 4);  // N=64  -> 4 n-tiles

    const size_t aggBytes = (size_t)NN * IN_DIM * sizeof(float);
    const int spmmBlocks = (NE * 32 + 255) / 256;

    // 2) agg = A @ x
    hipMemsetAsync(aggF, 0, aggBytes, stream);
    gcn_spmm<<<spmmBlocks, 256, 0, stream>>>(row, col, vals, x, aggF, NE);

    // 3) h = relu(agg @ W1): one wave per 16-row strip, 8 n-tiles per wave
    {
        int mtiles = NN / 16;                 // 3125
        int blocks = (mtiles + 3) / 4;        // 4 waves per block
        gcn_gemm_wmma<8, true><<<blocks, 128, 0, stream>>>(aggF, W1p, hF, NN);
    }

    // 4) agg = A @ h
    hipMemsetAsync(aggF, 0, aggBytes, stream);
    gcn_spmm<<<spmmBlocks, 256, 0, stream>>>(row, col, vals, hF, aggF, NE);

    // 5) out = agg @ W2: 4 n-tiles per wave
    {
        int mtiles = NN / 16;
        int blocks = (mtiles + 3) / 4;
        gcn_gemm_wmma<4, false><<<blocks, 128, 0, stream>>>(aggF, W2p, out, NN);
    }
}
